// MinimalHGRNCore_1546188227052
// MI455X (gfx1250) — 8-bit matrix operand order repaired and checked by reading, compiled, not run on hardware
//
#include <hip/hip_runtime.h>
#include <stdint.h>


typedef __attribute__((ext_vector_type(8))) int          v8i_t;
typedef __attribute__((ext_vector_type(4))) int          v4i_t;
typedef __attribute__((ext_vector_type(4))) unsigned int v4u_t;
typedef __attribute__((ext_vector_type(2))) unsigned int v2u_t;

union FragA { v8i_t v; v2u_t q[4]; };
union FragB { v8i_t v; v4u_t u4[2]; };

#if __has_builtin(__builtin_amdgcn_s_wait_asynccnt)
#define WAIT_ASYNC(n) __builtin_amdgcn_s_wait_asynccnt(n)
#else
#define WAIT_ASYNC(n) asm volatile("s_wait_asynccnt %0" ::"n"(n) : "memory")
#endif

#if __has_builtin(__builtin_amdgcn_s_wait_dscnt)
#define WAIT_DS0() __builtin_amdgcn_s_wait_dscnt(0)
#else
#define WAIT_DS0() asm volatile("s_wait_dscnt 0x0" ::: "memory")
#endif

#if __has_builtin(__builtin_amdgcn_s_barrier_signal) && \
    __has_builtin(__builtin_amdgcn_s_barrier_wait)
#define WG_BARRIER()                        \
  do {                                      \
    __builtin_amdgcn_s_barrier_signal(-1);  \
    __builtin_amdgcn_s_barrier_wait(-1);    \
  } while (0)
#else
#define WG_BARRIER() __syncthreads()
#endif

__device__ __forceinline__ void fill_b128(char* dst_lds, const int8_t* src_g) {
#if __has_builtin(__builtin_amdgcn_global_load_async_to_lds_b128)
  typedef v4i_t __attribute__((address_space(1))) * gp128_t;
  typedef v4i_t __attribute__((address_space(3))) * lp128_t;
  __builtin_amdgcn_global_load_async_to_lds_b128((gp128_t)src_g,
                                                 (lp128_t)dst_lds, 0, 0);
#else
  unsigned loff =
      (unsigned)(uintptr_t)(__attribute__((address_space(3))) char*)dst_lds;
  asm volatile("global_load_async_to_lds_b128 %0, %1, off" ::"v"(loff),
               "v"(src_g)
               : "memory");
#endif
}

__device__ __forceinline__ float blk_reduce_sum(float v, float* red) {
  int t = threadIdx.x;
  red[t] = v; __syncthreads();
  for (int s = blockDim.x >> 1; s > 0; s >>= 1) {
    if (t < s) red[t] += red[t + s];
    __syncthreads();
  }
  float r = red[0]; __syncthreads();
  return r;
}
__device__ __forceinline__ float blk_reduce_max(float v, float* red) {
  int t = threadIdx.x;
  red[t] = v; __syncthreads();
  for (int s = blockDim.x >> 1; s > 0; s >>= 1) {
    if (t < s) red[t] = fmaxf(red[t], red[t + s]);
    __syncthreads();
  }
  float r = red[0]; __syncthreads();
  return r;
}

__global__ void wabs_sum_kernel(const float* __restrict__ w,
                                float* __restrict__ accum, int n) {
  __shared__ float red[256];
  float s = 0.f;
  for (int i = blockIdx.x * blockDim.x + threadIdx.x; i < n;
       i += gridDim.x * blockDim.x)
    s += fabsf(w[i]);
  s = blk_reduce_sum(s, red);
  if (threadIdx.x == 0) atomicAdd(accum, s);
}

__global__ void wquant_kernel(const float* __restrict__ w,
                              const float* __restrict__ accum,
                              int8_t* __restrict__ wq, int n) {
  float mean  = fmaxf(accum[0] / (float)n, 1e-5f);
  float scale = 1.f / mean;
  for (int i = blockIdx.x * blockDim.x + threadIdx.x; i < n;
       i += gridDim.x * blockDim.x) {
    float q = rintf(w[i] * scale);
    q = fminf(fmaxf(q, -1.f), 1.f);
    wq[i] = (int8_t)q;
  }
}

__global__ void rms_quant3_kernel(const float* __restrict__ x,
                                  const float* __restrict__ wi,
                                  const float* __restrict__ wf,
                                  const float* __restrict__ wg,
                                  int8_t* __restrict__ Ai, int8_t* __restrict__ Af,
                                  int8_t* __restrict__ Ag,
                                  float* __restrict__ dqi, float* __restrict__ dqf,
                                  float* __restrict__ dqg, int H) {
  extern __shared__ float smem[];
  float* xs  = smem;
  float* red = smem + H;
  const size_t row = blockIdx.x;
  const float* xp = x + row * (size_t)H;

  float ss = 0.f;
  for (int h = threadIdx.x; h < H; h += blockDim.x) {
    float v = xp[h]; xs[h] = v; ss += v * v;
  }
  ss = blk_reduce_sum(ss, red);
  const float rstd = rsqrtf(ss / (float)H + 1e-8f);

  const float* nw[3] = {wi, wf, wg};
  int8_t* A[3]       = {Ai, Af, Ag};
  float*  dq[3]      = {dqi, dqf, dqg};
  for (int c = 0; c < 3; ++c) {
    float mx = 0.f;
    for (int h = threadIdx.x; h < H; h += blockDim.x)
      mx = fmaxf(mx, fabsf(xs[h] * rstd * nw[c][h]));
    mx = blk_reduce_max(mx, red);
    mx = fmaxf(mx, 1e-5f);
    const float scale = 127.f / mx;
    for (int h = threadIdx.x; h < H; h += blockDim.x) {
      float q = rintf(xs[h] * rstd * nw[c][h] * scale);
      q = fminf(fmaxf(q, -128.f), 127.f);
      A[c][row * (size_t)H + h] = (int8_t)q;
    }
    if (threadIdx.x == 0) dq[c][row] = mx / 127.f;
  }
}

#define BSTRIDE 80
#define BSTAGE  (64 * BSTRIDE)

__device__ __forceinline__ void load_a(FragA& fa, const int8_t* p) {
  const v2u_t* pa = (const v2u_t*)p;
  fa.q[0] = pa[0]; fa.q[1] = pa[1]; fa.q[2] = pa[4]; fa.q[3] = pa[5];
}

__global__ void __launch_bounds__(256)
gemm_i8_wmma_kernel(const int8_t* __restrict__ A8, const float* __restrict__ dqA,
                    const int8_t* __restrict__ Wq, const float* __restrict__ wsum,
                    float inv_wcount, float* __restrict__ out,
                    int M, int N, int K) {
  __shared__ __attribute__((aligned(16))) char bsmem[3 * BSTAGE];

  const int tid  = threadIdx.x;
  const int lane = tid & 31;
  const int wave = tid >> 5;
  const int hi   = lane >> 4;
  const int lm   = lane & 15;
  const int m0   = blockIdx.y * 128 + wave * 16;
  const int n0   = blockIdx.x * 64;

  const int fn = tid >> 2;
  const int fc = tid & 3;
  const int8_t* gB   = Wq + (size_t)(n0 + fn) * (size_t)K + fc * 16;
  char*         ldsB = bsmem + fn * BSTRIDE + fc * 16;

  const size_t arow  = (size_t)(m0 + lm) * (size_t)K;
  const int    STEPS = K >> 6;

  v8i_t acc[4] = {};
  FragA fa0, fa1;

  fill_b128(ldsB + 0, gB + 0);
  if (STEPS > 1) fill_b128(ldsB + BSTAGE, gB + 64);
  load_a(fa0, A8 + arow + (hi << 4));

  unsigned rd   = 0;
  unsigned fill = 2u * BSTAGE;

  auto compute = [&](FragA& cur) {
    const char* bb = bsmem + rd;
    FragB fb[4];
#pragma unroll
    for (int j = 0; j < 4; ++j) {
      const char* rp = bb + (j * 16 + lm) * BSTRIDE + (hi << 4);
      fb[j].u4[0] = *(const v4u_t*)(rp);
      fb[j].u4[1] = *(const v4u_t*)(rp + 32);
    }
    const v8i_t av = cur.v;
#pragma unroll
    for (int j = 0; j < 4; ++j)
      acc[j] = __builtin_amdgcn_wmma_i32_16x16x64_iu8(
           true, av,  true, fb[j].v, acc[j],
           false,  false);
  };
  auto advance = [&]() {
    fill = rd;
    rd   = (rd == 2u * BSTAGE) ? 0u : rd + BSTAGE;
  };

  int it = 0;
  for (; it + 3 < STEPS; it += 2) {
    WAIT_ASYNC(1); WAIT_DS0(); WG_BARRIER();
    fill_b128(ldsB + fill, gB + (it << 6) + 128);
    load_a(fa1, A8 + arow + (it << 6) + 64 + (hi << 4));
    compute(fa0); advance();

    WAIT_ASYNC(1); WAIT_DS0(); WG_BARRIER();
    fill_b128(ldsB + fill, gB + ((it + 1) << 6) + 128);
    load_a(fa0, A8 + arow + ((it + 1) << 6) + 64 + (hi << 4));
    compute(fa1); advance();
  }
  auto tail = [&](int t, FragA& cur, FragA& nxt) {
    if (t + 1 < STEPS) { WAIT_ASYNC(1); } else { WAIT_ASYNC(0); }
    WAIT_DS0(); WG_BARRIER();
    if (t + 2 < STEPS) fill_b128(ldsB + fill, gB + (t << 6) + 128);
    if (t + 1 < STEPS) load_a(nxt, A8 + arow + (t << 6) + 64 + (hi << 4));
    compute(cur); advance();
  };
  if (it < STEPS) { tail(it, fa0, fa1); ++it; }
  if (it < STEPS) { tail(it, fa1, fa0); ++it; }
  if (it < STEPS) { tail(it, fa0, fa1); ++it; }

  const float wdq = fmaxf(wsum[0] * inv_wcount, 1e-5f);
#pragma unroll
  for (int j = 0; j < 4; ++j) {
    const int n = n0 + j * 16 + lm;
#pragma unroll
    for (int r = 0; r < 8; ++r) {
      const int m = m0 + r + (hi << 3);
      out[(size_t)m * N + n] = (float)acc[j][r] * dqA[m] * wdq;
    }
  }
}

__global__ void scan_local_kernel(const float* __restrict__ i_raw,
                                  float* __restrict__ f_raw,
                                  float* __restrict__ cumf,
                                  float* __restrict__ chP, float* __restrict__ chS,
                                  int B, int L, int E, int C) {
  int idx = blockIdx.x * blockDim.x + threadIdx.x;
  if (idx >= B * E * C) return;
  const int be = idx % (B * E);
  const int c  = idx / (B * E);
  const int b  = be / E, e = be % E;
  const int Lc = L / C;
  const size_t base = ((size_t)b * L + (size_t)c * Lc) * E + e;
  float s = 0.f, p = 1.f;
  for (int t = 0; t < Lc; ++t) {
    const size_t q = base + (size_t)t * E;
    const float fl = f_raw[q];
    const float il = i_raw[q];
    const float ft = 1.f / (1.f + __expf(-fl));
    const float it = il * (1.f / (1.f + __expf(-il))) * (1.f - ft);
    s = ft * s + it;
    p *= ft;
    f_raw[q] = s;
    cumf[q]  = p;
  }
  chP[idx] = p;
  chS[idx] = s;
}

__global__ void scan_prefix_kernel(const float* __restrict__ chP,
                                   const float* __restrict__ chS,
                                   float* __restrict__ pre, int BE, int C) {
  int be = blockIdx.x * blockDim.x + threadIdx.x;
  if (be >= BE) return;
  float s0 = 0.f;
  for (int c = 0; c < C; ++c) {
    pre[(size_t)c * BE + be] = s0;
    s0 = chS[(size_t)c * BE + be] + s0 * chP[(size_t)c * BE + be];
  }
}

__global__ void scan_apply_kernel(float* __restrict__ f_raw,
                                  const float* __restrict__ cumf,
                                  const float* __restrict__ pre,
                                  int B, int L, int E, int C) {
  const size_t n = (size_t)B * L * E;
  const size_t i = (size_t)blockIdx.x * blockDim.x + threadIdx.x;
  if (i >= n) return;
  const int e  = (int)(i % E);
  const size_t bt = i / E;
  const int t  = (int)(bt % L);
  const int b  = (int)(bt / L);
  const int c  = t / (L / C);
  const float s0 = pre[(size_t)c * (B * E) + (size_t)b * E + e];
  f_raw[i] += s0 * cumf[i];
}

__global__ void out_norm_quant_kernel(const float* __restrict__ g_raw,
                                      const float* __restrict__ s_all,
                                      const float* __restrict__ rms_w,
                                      const float* __restrict__ norm_o,
                                      int8_t* __restrict__ Ao,
                                      float* __restrict__ dqo, int E) {
  extern __shared__ float smem[];
  float* ys  = smem;
  float* red = smem + E;
  const size_t row = blockIdx.x;
  const float* sp = s_all + row * (size_t)E;
  const float* gp = g_raw + row * (size_t)E;

  float ss = 0.f;
  for (int h = threadIdx.x; h < E; h += blockDim.x) {
    float v = sp[h]; ys[h] = v; ss += v * v;
  }
  ss = blk_reduce_sum(ss, red);
  const float rstd_s = rsqrtf(ss / (float)E + 1e-5f);

  float ss2 = 0.f;
  for (int h = threadIdx.x; h < E; h += blockDim.x) {
    const float gl = gp[h];
    const float y  = gl * (1.f / (1.f + __expf(-gl))) *
                     (ys[h] * rstd_s * rms_w[h]);
    ys[h] = y;
    ss2 += y * y;
  }
  ss2 = blk_reduce_sum(ss2, red);
  const float rstd_y = rsqrtf(ss2 / (float)E + 1e-8f);

  float mx = 0.f;
  for (int h = threadIdx.x; h < E; h += blockDim.x)
    mx = fmaxf(mx, fabsf(ys[h] * rstd_y * norm_o[h]));
  mx = blk_reduce_max(mx, red);
  mx = fmaxf(mx, 1e-5f);
  const float scale = 127.f / mx;
  for (int h = threadIdx.x; h < E; h += blockDim.x) {
    float q = rintf(ys[h] * rstd_y * norm_o[h] * scale);
    q = fminf(fmaxf(q, -128.f), 127.f);
    Ao[row * (size_t)E + h] = (int8_t)q;
  }
  if (threadIdx.x == 0) dqo[row] = mx / 127.f;
}

extern "C" void kernel_launch(void* const* d_in, const int* in_sizes, int n_in,
                              void* d_out, int out_size, void* d_ws,
                              size_t ws_size, hipStream_t stream) {
  const float* x      = (const float*)d_in[0];
  const float* Wi     = (const float*)d_in[1];
  const float* Wf     = (const float*)d_in[2];
  const float* Wg     = (const float*)d_in[3];
  const float* Wo     = (const float*)d_in[4];
  const float* norm_i = (const float*)d_in[5];
  const float* norm_f = (const float*)d_in[6];
  const float* norm_g = (const float*)d_in[7];
  const float* norm_o = (const float*)d_in[8];
  const float* rms_w  = (const float*)d_in[9];

  const int H  = in_sizes[5];
  const int E  = in_sizes[1] / H;
  const int M  = in_sizes[0] / H;
  const int B  = 4;
  const int L  = M / B;
  const int nW = E * H;
  int C = 16;
  while (C > 1 && (L % C)) C >>= 1;

  char*  ws  = (char*)d_ws;
  size_t off = 0;
  auto alloc = [&](size_t bytes) -> void* {
    void* p = ws + off;
    off += (bytes + 255) & ~(size_t)255;
    return p;
  };
  float*  wsum  = (float*)alloc(4 * sizeof(float));
  int8_t* wq_i  = (int8_t*)alloc((size_t)nW);
  int8_t* wq_f  = (int8_t*)alloc((size_t)nW);
  int8_t* wq_g  = (int8_t*)alloc((size_t)nW);
  int8_t* wq_o  = (int8_t*)alloc((size_t)nW);
  int8_t* Ai    = (int8_t*)alloc((size_t)M * H);
  int8_t* Af    = (int8_t*)alloc((size_t)M * H);
  int8_t* Ag    = (int8_t*)alloc((size_t)M * H);
  float*  dqi   = (float*)alloc((size_t)M * sizeof(float));
  float*  dqf   = (float*)alloc((size_t)M * sizeof(float));
  float*  dqg   = (float*)alloc((size_t)M * sizeof(float));
  float*  i_raw = (float*)alloc((size_t)M * E * sizeof(float));
  float*  f_raw = (float*)alloc((size_t)M * E * sizeof(float));
  float*  g_raw = (float*)alloc((size_t)M * E * sizeof(float));
  float*  cumf  = (float*)alloc((size_t)M * E * sizeof(float));
  float*  chP   = (float*)alloc((size_t)B * E * C * sizeof(float));
  float*  chS   = (float*)alloc((size_t)B * E * C * sizeof(float));
  float*  pre   = (float*)alloc((size_t)B * E * C * sizeof(float));

  (void)hipMemsetAsync(wsum, 0, 4 * sizeof(float), stream);
  wabs_sum_kernel<<<256, 256, 0, stream>>>(Wi, wsum + 0, nW);
  wabs_sum_kernel<<<256, 256, 0, stream>>>(Wf, wsum + 1, nW);
  wabs_sum_kernel<<<256, 256, 0, stream>>>(Wg, wsum + 2, nW);
  wabs_sum_kernel<<<256, 256, 0, stream>>>(Wo, wsum + 3, nW);
  wquant_kernel<<<512, 256, 0, stream>>>(Wi, wsum + 0, wq_i, nW);
  wquant_kernel<<<512, 256, 0, stream>>>(Wf, wsum + 1, wq_f, nW);
  wquant_kernel<<<512, 256, 0, stream>>>(Wg, wsum + 2, wq_g, nW);
  wquant_kernel<<<512, 256, 0, stream>>>(Wo, wsum + 3, wq_o, nW);

  const size_t sh_q3 = (size_t)(H + 256) * sizeof(float);
  rms_quant3_kernel<<<M, 256, sh_q3, stream>>>(x, norm_i, norm_f, norm_g, Ai,
                                               Af, Ag, dqi, dqf, dqg, H);

  dim3 gemm_grid(E / 64, M / 128);
  const float invW = 1.f / (float)nW;
  gemm_i8_wmma_kernel<<<gemm_grid, 256, 0, stream>>>(Ai, dqi, wq_i, wsum + 0,
                                                     invW, i_raw, M, E, H);
  gemm_i8_wmma_kernel<<<gemm_grid, 256, 0, stream>>>(Af, dqf, wq_f, wsum + 1,
                                                     invW, f_raw, M, E, H);
  gemm_i8_wmma_kernel<<<gemm_grid, 256, 0, stream>>>(Ag, dqg, wq_g, wsum + 2,
                                                     invW, g_raw, M, E, H);

  const int BE = B * E;
  scan_local_kernel<<<(BE * C + 255) / 256, 256, 0, stream>>>(
      i_raw, f_raw, cumf, chP, chS, B, L, E, C);
  scan_prefix_kernel<<<(BE + 255) / 256, 256, 0, stream>>>(chP, chS, pre, BE, C);
  {
    const size_t n = (size_t)M * E;
    scan_apply_kernel<<<(unsigned)((n + 255) / 256), 256, 0, stream>>>(
        f_raw, cumf, pre, B, L, E, C);
  }

  const size_t sh_o = (size_t)(E + 256) * sizeof(float);
  out_norm_quant_kernel<<<M, 256, sh_o, stream>>>(g_raw, f_raw, rms_w, norm_o,
                                                  Ai, dqi, E);

  dim3 gemm_grid_o(H / 64, M / 128);
  gemm_i8_wmma_kernel<<<gemm_grid_o, 256, 0, stream>>>(
      Ai, dqi, wq_o, wsum + 3, invW, (float*)d_out, M, H, E);
}
